// DynamicModuleHead_87041807221282
// MI455X (gfx1250) — compile-verified, matrix operand order checked by reading, not run on hardware
//
#include <hip/hip_runtime.h>
#include <math.h>
#include <stdint.h>

typedef __attribute__((ext_vector_type(8)))  float   v8f;
typedef __attribute__((ext_vector_type(8)))  __bf16  v8bf;
typedef __attribute__((ext_vector_type(16))) __bf16  v16bf;
typedef __attribute__((ext_vector_type(4)))  unsigned int u32x4;
typedef __attribute__((ext_vector_type(8)))  int     i32x8;
typedef __attribute__((ext_vector_type(4)))  int     i32x4;

#define HH   128
#define WW   128
#define CCH  256
#define NPIX (HH*WW)      /* 16384 */
#define KCH  1536         /* concat channels */
#define KOFF 2304         /* 9*256, offset-conv K */

// ---------------- fragment helpers (bf16, 16x16x32 WMMA) ----------------
// Row-major M x K (or N x K for B^T): lane L holds row L&15,
// halves K = kb..kb+7 and kb+16..kb+23 with kb = (L>=16)?8:0.
__device__ __forceinline__ v16bf frag16(const __bf16* base, int ldk, int k0) {
  int lane = threadIdx.x & 31;
  int row  = lane & 15;
  int kb   = (lane >> 4) << 3;
  const __bf16* p = base + row * ldk + k0 + kb;
  v8bf lo = *(const v8bf*)(p);
  v8bf hi = *(const v8bf*)(p + 16);
  return __builtin_shufflevector(lo, hi, 0,1,2,3,4,5,6,7,8,9,10,11,12,13,14,15);
}

// B^T fragment for the offset conv: pixels of one row, tap-shifted column,
// per-lane zero outside the image (reconverges before the WMMA).
__device__ __forceinline__ v16bf frag_b_sample(const __bf16* catbf, int hy,
                                               int w0, int dxo, int c0) {
  int lane = threadIdx.x & 31;
  int wx   = w0 + (lane & 15) + dxo;
  int kb   = (lane >> 4) << 3;
  v16bf z = {};
  if (wx < 0 || wx >= WW) return z;
  const __bf16* p = catbf + (hy * WW + wx) * KCH + c0 + kb;
  v8bf lo = *(const v8bf*)(p);
  v8bf hi = *(const v8bf*)(p + 16);
  return __builtin_shufflevector(lo, hi, 0,1,2,3,4,5,6,7,8,9,10,11,12,13,14,15);
}

// ---------------- prep kernels ----------------
__global__ void k_zero(float* p, int n) {
  int i = blockIdx.x * blockDim.x + threadIdx.x;
  if (i < n) p[i] = 0.f;
}

// x (C,H,W) f32 -> catbf pixel-major (p, c) bf16, channels 0..255
__global__ void k_cvt_x(const float* __restrict__ x, __bf16* __restrict__ catbf) {
  int id = blockIdx.x * 256 + threadIdx.x;   // id < C*NPIX
  int p  = id >> 8;
  int c  = id & 255;
  catbf[p * KCH + c] = (__bf16)x[c * NPIX + p];
}

// w (27,256,3,3) f32 -> A-matrix 32 x 2304 bf16, K = tap*256 + c, rows 27..31 zero
__global__ void k_cvt_w(const float* __restrict__ w, __bf16* __restrict__ wbfb) {
  int id = blockIdx.x * 256 + threadIdx.x;   // id < 32*2304
  int m = id / KOFF, k = id % KOFF;
  int t = k >> 8, c = k & 255;
  float v = (m < 27) ? w[(m * CCH + c) * 9 + t] : 0.f;
  wbfb[id] = (__bf16)v;
}

// w_scale (256,1536) f32 -> bf16 row-major
__global__ void k_cvt_ws(const float* __restrict__ w, __bf16* __restrict__ wsbf) {
  int id = blockIdx.x * 256 + threadIdx.x;   // id < 256*1536
  wsbf[id] = (__bf16)w[id];
}

// ---------------- offset conv: 5 branches, M=32(27) x K=2304 x N=16384 ----------------
__global__ void k_offconv(const __bf16* __restrict__ catbf,
                          const __bf16* __restrict__ wbf,
                          float* __restrict__ dfo) {
  const int dils[5] = {1, 6, 12, 24, 36};
  int b = blockIdx.y;
  int d = dils[b];
  int wave = threadIdx.x >> 5;
  int p0   = (blockIdx.x * 4 + wave) * 16;   // 16-pixel N tile (never crosses a row)
  int h    = p0 >> 7, w0 = p0 & 127;
  const __bf16* A = wbf + b * 32 * KOFF;

  v8f acc0 = {};   // rows 0..15
  v8f acc1 = {};   // rows 16..31
  for (int t = 0; t < 9; ++t) {
    int ky = t / 3, kx = t % 3;
    int hy = h + (ky - 1) * d;               // wave-uniform
    if (hy < 0 || hy >= HH) continue;
    int dxo = (kx - 1) * d;
    #pragma unroll
    for (int cc = 0; cc < 8; ++cc) {
      int k0 = t * 256 + cc * 32;
      v16bf bf = frag_b_sample(catbf, hy, w0, dxo, cc * 32);
      v16bf a0 = frag16(A, KOFF, k0);
      v16bf a1 = frag16(A + 16 * KOFF, KOFF, k0);
      acc0 = __builtin_amdgcn_wmma_f32_16x16x32_bf16(false, a0, false, bf, (short)0, acc0, false, false);
      acc1 = __builtin_amdgcn_wmma_f32_16x16x32_bf16(false, a1, false, bf, (short)0, acc1, false, false);
    }
  }
  int lane = threadIdx.x & 31;
  int p    = p0 + (lane & 15);
  int mh   = (lane >> 4) << 3;
  #pragma unroll
  for (int j = 0; j < 8; ++j) {
    int m0 = j + mh;                                   // 0..15, always < 27
    dfo[(b * 27 + m0) * NPIX + p] = acc0[j];
    int m1 = 16 + j + mh;
    if (m1 < 27) dfo[(b * 27 + m1) * NPIX + p] = acc1[j];
  }
}

// ---------------- deformable softmax-weighted bilinear gather ----------------
__global__ void k_deform(__bf16* catbf, const float* __restrict__ dfo) {
  const int dils[5] = {1, 6, 12, 24, 36};
  int b = blockIdx.y;
  int d = dils[b];
  int wave = threadIdx.x >> 5;
  int lane = threadIdx.x & 31;
  int p = blockIdx.x * 8 + wave;             // one pixel per wave
  int y = p >> 7, xw = p & 127;
  const float* F = dfo + b * 27 * NPIX;

  float f[9]; float mx = -1e30f;
  #pragma unroll
  for (int k = 0; k < 9; ++k) { f[k] = F[k * NPIX + p]; mx = fmaxf(mx, f[k]); }
  float s = 0.f;
  #pragma unroll
  for (int k = 0; k < 9; ++k) { f[k] = __expf(f[k] - mx); s += f[k]; }
  float inv = 1.f / s;

  float acc[8];
  #pragma unroll
  for (int i = 0; i < 8; ++i) acc[i] = 0.f;

  #pragma unroll
  for (int k = 0; k < 9; ++k) {
    int ky = k / 3, kx = k % 3;
    float dy = F[(9 + 2 * k)  * NPIX + p];
    float dx = F[(10 + 2 * k) * NPIX + p];
    float py = (float)y  + (float)((ky - 1) * d) + dy;
    float px = (float)xw + (float)((kx - 1) * d) + dx;
    float y0f = floorf(py), x0f = floorf(px);
    int y0 = (int)y0f, x0 = (int)x0f;
    int y1 = y0 + 1,   x1 = x0 + 1;
    float wy1 = py - y0f, wy0 = 1.f - wy1;
    float wx1 = px - x0f, wx0 = 1.f - wx1;
    float fk = f[k] * inv;
    float vy0 = (y0 >= 0 && y0 < HH) ? 1.f : 0.f;
    float vy1 = (y1 >= 0 && y1 < HH) ? 1.f : 0.f;
    float vx0 = (x0 >= 0 && x0 < WW) ? 1.f : 0.f;
    float vx1 = (x1 >= 0 && x1 < WW) ? 1.f : 0.f;
    int y0c = min(max(y0, 0), HH - 1), y1c = min(max(y1, 0), HH - 1);
    int x0c = min(max(x0, 0), WW - 1), x1c = min(max(x1, 0), WW - 1);
    int   cb[4]; float cw[4];
    cb[0] = (y0c * WW + x0c) * KCH;  cw[0] = fk * wy0 * wx0 * vy0 * vx0;
    cb[1] = (y0c * WW + x1c) * KCH;  cw[1] = fk * wy0 * wx1 * vy0 * vx1;
    cb[2] = (y1c * WW + x0c) * KCH;  cw[2] = fk * wy1 * wx0 * vy1 * vx0;
    cb[3] = (y1c * WW + x1c) * KCH;  cw[3] = fk * wy1 * wx1 * vy1 * vx1;
    #pragma unroll
    for (int j = 0; j < 4; ++j) {
      const __bf16* src = catbf + cb[j] + lane;   // channels 0..255 = x
      float wj = cw[j];
      #pragma unroll
      for (int i = 0; i < 8; ++i)
        acc[i] += wj * (float)src[32 * i];
    }
  }
  __bf16* outp = catbf + p * KCH + 256 + b * CCH;
  #pragma unroll
  for (int i = 0; i < 8; ++i) outp[lane + 32 * i] = (__bf16)acc[i];
}

// ---------------- 1x1 conv GEMM: M=256, K=1536, N=16384 + BN partial sums ----------
// B tile (16 pixels x 1536 ch = one contiguous 48 KB span of catbf) is DMA'd
// into LDS once per block by the Tensor Data Mover; all 4 waves then read
// B fragments from LDS (ds_load_b128), A fragments stay L2-resident.
__global__ void k_gemm1x1(const __bf16* __restrict__ catbf,
                          const __bf16* __restrict__ wsbf,
                          float* __restrict__ out, float* __restrict__ bnacc) {
  __shared__ __bf16 btile[16 * KCH];         // 49152 B
  int wave = threadIdx.x >> 5;
  int lane = threadIdx.x & 31;
  int p0 = blockIdx.x * 16;
  int mbase = wave * 64;                     // 4 waves cover M = 0..255

  if (wave == 0) {
    // 1-D TDM descriptor: 6144 units of 8 B = 48 KB, contiguous.
    unsigned long long ga = (unsigned long long)(uintptr_t)(catbf + (size_t)p0 * KCH);
    unsigned int la = (unsigned int)(uintptr_t)(&btile[0]);
    u32x4 g0;
    g0.x = 1u;                                            // count=1, user mode
    g0.y = la;                                            // lds_addr
    g0.z = (unsigned int)(ga & 0xffffffffu);              // global_addr[31:0]
    g0.w = (unsigned int)((ga >> 32) & 0x01ffffffu) | (2u << 30);  // [56:32] | type=2
    i32x8 g1;
    g1[0] = 0x00030000;                                   // data_size = 3 (8 B)
    g1[1] = (int)((6144u & 0xffffu) << 16);               // tensor_dim0[15:0] @ bit48
    g1[2] = 0;                                            // tensor_dim0 hi, tensor_dim1 lo
    g1[3] = (int)(6144u << 16);                           // tile_dim0 @ bit112
    g1[4] = 0;                                            // tile_dim1/2 = 0 (1-D)
    g1[5] = 6144;                                         // tensor_dim0_stride lo
    g1[6] = 0;
    g1[7] = 0;
    i32x4 z4 = {0, 0, 0, 0};
#if __has_include(<hip/amd_detail/amd_gfx1250_TDM.h>)
    i32x8 z8 = {};
    __builtin_amdgcn_tensor_load_to_lds(g0, g1, z4, z4, z8, 0);   // 6-arg toolchain
#else
    __builtin_amdgcn_tensor_load_to_lds(g0, g1, z4, z4, 0);       // 5-arg (ROCm 7.2)
#endif
    __builtin_amdgcn_s_wait_tensorcnt(0);
  }
  __syncthreads();

  v8f acc[4] = {{}, {}, {}, {}};
  for (int k = 0; k < KCH; k += 32) {
    v16bf bf = frag16((const __bf16*)btile, KCH, k);      // ds_load_b128 x2
    #pragma unroll
    for (int g = 0; g < 4; ++g) {
      v16bf af = frag16(wsbf + (mbase + g * 16) * KCH, KCH, k);
      acc[g] = __builtin_amdgcn_wmma_f32_16x16x32_bf16(false, af, false, bf, (short)0, acc[g], false, false);
    }
  }
  int mh = (lane >> 4) << 3;
  int p  = p0 + (lane & 15);
  #pragma unroll
  for (int g = 0; g < 4; ++g) {
    #pragma unroll
    for (int j = 0; j < 8; ++j) {
      int m = mbase + g * 16 + j + mh;
      float v = acc[g][j];
      out[m * NPIX + p] = v;
      // lanes {0..15} (and {16..31}) of this VGPR share one channel -> 16-lane tree
      float s = v, s2 = v * v;
      #pragma unroll
      for (int off = 1; off < 16; off <<= 1) {
        s  += __shfl_xor(s,  off);
        s2 += __shfl_xor(s2, off);
      }
      if ((lane & 15) == 0) {
        atomicAdd(&bnacc[m], s);
        atomicAdd(&bnacc[256 + m], s2);
      }
    }
  }
}

// ---------------- BN normalize (population stats) ----------------
__global__ void k_bn(float* __restrict__ out, const float* __restrict__ bnacc,
                     const float* __restrict__ gamma, const float* __restrict__ beta) {
  int id = blockIdx.x * 256 + threadIdx.x;   // id < C*NPIX
  int m = id >> 14;                          // /NPIX
  float mean = bnacc[m] * (1.f / NPIX);
  float var  = bnacc[256 + m] * (1.f / NPIX) - mean * mean;
  float v = out[id];
  out[id] = (v - mean) * rsqrtf(var + 1e-5f) * gamma[m] + beta[m];
}

// ---------------- host ----------------
extern "C" void kernel_launch(void* const* d_in, const int* in_sizes, int n_in,
                              void* d_out, int out_size, void* d_ws, size_t ws_size,
                              hipStream_t stream) {
  (void)in_sizes; (void)n_in; (void)out_size; (void)ws_size;
  const float* x       = (const float*)d_in[0];
  const float* wconv[5] = {(const float*)d_in[1], (const float*)d_in[2],
                           (const float*)d_in[3], (const float*)d_in[4],
                           (const float*)d_in[5]};
  const float* w_scale = (const float*)d_in[6];
  const float* gamma   = (const float*)d_in[7];
  const float* beta    = (const float*)d_in[8];
  float* out = (float*)d_out;

  // workspace layout (bytes)
  char* ws = (char*)d_ws;
  __bf16* catbf = (__bf16*)(ws);                         // 16384*1536*2 = 50,331,648
  __bf16* wbf   = (__bf16*)(ws + 50331648);              // 5*32*2304*2  =    737,280
  __bf16* wsbf  = (__bf16*)(ws + 50331648 + 737280);     // 256*1536*2   =    786,432
  float*  dfo   = (float*) (ws + 51855360);              // 5*27*16384*4 =  8,847,360
  float*  bnacc = (float*) (ws + 60702720);              // 512*4        =      2,048

  k_zero<<<2, 256, 0, stream>>>(bnacc, 512);
  k_cvt_x<<<(CCH * NPIX) / 256, 256, 0, stream>>>(x, catbf);
  for (int b = 0; b < 5; ++b)
    k_cvt_w<<<(32 * KOFF) / 256, 256, 0, stream>>>(wconv[b], wbf + b * 32 * KOFF);
  k_cvt_ws<<<(CCH * KCH) / 256, 256, 0, stream>>>(w_scale, wsbf);

  k_offconv<<<dim3(256, 5), 128, 0, stream>>>(catbf, wbf, dfo);
  k_deform <<<dim3(2048, 5), 256, 0, stream>>>(catbf, dfo);
  k_gemm1x1<<<1024, 128, 0, stream>>>(catbf, wsbf, out, bnacc);
  k_bn     <<<(CCH * NPIX) / 256, 256, 0, stream>>>(out, bnacc, gamma, beta);
}
